// HybridQuanvolutionNet_65481071397256
// MI455X (gfx1250) — matrix operand order repaired and checked by reading, not compiled, not run on hardware
//
#include <hip/hip_runtime.h>

typedef __attribute__((ext_vector_type(16))) _Float16 v16h;
typedef __attribute__((ext_vector_type(8)))  float    v8f;

// ---------------------------------------------------------------------------
// Build the A-operand (16x32 f16) for a complex 16x16 unitary applied to a
// state with real part in K=0..15 and imag part in K=16..31:
//   A_re = [Ur | -Ui],  A_im = [Ui | Ur]
// Per-lane K map (contiguous, matching the B operand):
//   lanes 0-15 : K=e      (K 0..15)
//   lanes 16-31: K=16+e   (K 16..31)
// ---------------------------------------------------------------------------
__device__ __forceinline__ void build_A(const float* __restrict__ ure,
                                        const float* __restrict__ uim,
                                        int m, bool hi, v16h& a_re, v16h& a_im)
{
#pragma unroll
    for (int e = 0; e < 16; ++e) {
        int k = hi ? 16 + e : e;
        if (k < 16) {
            a_re[e] = (_Float16)ure[m * 16 + k];
            a_im[e] = (_Float16)uim[m * 16 + k];
        } else {
            int kk = k - 16;
            a_re[e] = (_Float16)(-uim[m * 16 + kk]);
            a_im[e] = (_Float16)(ure[m * 16 + kk]);
        }
    }
}

// ---------------------------------------------------------------------------
// One quantum layer for a 16-batch tile held by one wave:
//   RY product-state encode -> complex 16x16 unitary (2x v_wmma) -> <Z_w>.
// B operand: lane&15 = batch column; lanes 0-15 hold psi_real (K=e),
// lanes 16-31 hold psi_imag = 0 (K=16+e).
// D operand: lane holds rows 0-7 (lo) / 8-15 (hi) of its column; the Z
// reduction over all 16 rows is completed with a half-wave shfl_xor(16).
// ---------------------------------------------------------------------------
__device__ __forceinline__ void qlayer(const float ang[4],
                                       const v16h& a_re, const v16h& a_im,
                                       bool hi, bool valid,
                                       float out[4])
{
    float c[4], s[4];
#pragma unroll
    for (int w = 0; w < 4; ++w) {
        float h = 0.5f * ang[w];
        s[w] = __sinf(h);
        c[w] = __cosf(h);
    }
    float gate = (!hi && valid) ? 1.f : 0.f;  // imag rows + OOB columns = 0
    v16h bm;
#pragma unroll
    for (int e = 0; e < 16; ++e) {
        // wire 0 = most significant bit of the basis index
        float p = ((e & 8) ? s[0] : c[0]) * ((e & 4) ? s[1] : c[1]) *
                  ((e & 2) ? s[2] : c[2]) * ((e & 1) ? s[3] : c[3]);
        bm[e] = (_Float16)(p * gate);
    }

    v8f zero = {};
    v8f dre = __builtin_amdgcn_wmma_f32_16x16x32_f16(
        false, a_re, false, bm, (short)0, zero, false, false);
    v8f dim = __builtin_amdgcn_wmma_f32_16x16x32_f16(
        false, a_im, false, bm, (short)0, zero, false, false);

    float part[4] = {0.f, 0.f, 0.f, 0.f};
#pragma unroll
    for (int r = 0; r < 8; ++r) {
        int mm = hi ? (8 + r) : r;
        float p = dre[r] * dre[r] + dim[r] * dim[r];
#pragma unroll
        for (int w = 0; w < 4; ++w)
            part[w] += (((mm >> (3 - w)) & 1) ? -p : p);
    }
#pragma unroll
    for (int w = 0; w < 4; ++w)
        out[w] = part[w] + __shfl_xor(part[w], 16, 32);
}

// ---------------------------------------------------------------------------
// Kernel 1: per wave, 16 batch elements. Only patch 0 feeds the output
// (dead-code eliminate the other 195 patches). Writes m[B,4] to scratch.
// ---------------------------------------------------------------------------
__global__ __launch_bounds__(256) void quanv_qfc_kernel(
    const float* __restrict__ x,
    const float* __restrict__ uqf_re, const float* __restrict__ uqf_im,
    const float* __restrict__ uqc_re, const float* __restrict__ uqc_im,
    float* __restrict__ m_out, int B)
{
    int lane = threadIdx.x & 31;
    int wave = threadIdx.x >> 5;
    int tile = blockIdx.x * 8 + wave;
    int bb   = tile * 16;
    int m    = lane & 15;
    bool hi  = lane >= 16;
    int b    = bb + m;
    bool valid = b < B;

    v16h aqf_re, aqf_im, aqc_re, aqc_im;
    build_A(uqf_re, uqf_im, m, hi, aqf_re, aqf_im);
    build_A(uqc_re, uqc_im, m, hi, aqc_re, aqc_im);

    // patch 0 of the 28x28 image: x[b,0], x[b,1], x[b,28], x[b,29]
    const float* xb = x + (size_t)(valid ? b : 0) * 784;
    float ang[4];
    ang[0] = valid ? xb[0]  : 0.f;
    ang[1] = valid ? xb[1]  : 0.f;
    ang[2] = valid ? xb[28] : 0.f;
    ang[3] = valid ? xb[29] : 0.f;

    float f1[4], f2[4];
    qlayer(ang, aqf_re, aqf_im, hi, valid, f1);   // QuanvolutionFilter
    qlayer(f1,  aqc_re, aqc_im, hi, valid, f2);   // QFCModelQuantum

    if (!hi && valid) {
        float4 v = make_float4(f2[0], f2[1], f2[2], f2[3]);
        reinterpret_cast<float4*>(m_out)[b] = v;
    }
}

// ---------------------------------------------------------------------------
// Kernel 2: single block. Batch mean / biased var over m[B,4] (LDS tree),
// then BatchNorm + 10x4 classifier + log_softmax.
// ---------------------------------------------------------------------------
__global__ __launch_bounds__(256) void bn_head_kernel(
    const float* __restrict__ m4,
    const float* __restrict__ gamma, const float* __restrict__ beta,
    const float* __restrict__ wcls,  const float* __restrict__ bcls,
    float* __restrict__ out, int B)
{
    __shared__ float s_sum[256][4];
    __shared__ float s_sq[256][4];
    __shared__ float s_scale[4], s_shift[4];

    int tid = threadIdx.x;
    float sum[4] = {0.f, 0.f, 0.f, 0.f};
    float sq[4]  = {0.f, 0.f, 0.f, 0.f};
    for (int r = tid; r < B; r += 256) {
        float4 v = reinterpret_cast<const float4*>(m4)[r];
        float vv[4] = {v.x, v.y, v.z, v.w};
#pragma unroll
        for (int w = 0; w < 4; ++w) { sum[w] += vv[w]; sq[w] += vv[w] * vv[w]; }
    }
#pragma unroll
    for (int w = 0; w < 4; ++w) { s_sum[tid][w] = sum[w]; s_sq[tid][w] = sq[w]; }
    __syncthreads();
    for (int off = 128; off > 0; off >>= 1) {
        if (tid < off) {
#pragma unroll
            for (int w = 0; w < 4; ++w) {
                s_sum[tid][w] += s_sum[tid + off][w];
                s_sq[tid][w]  += s_sq[tid + off][w];
            }
        }
        __syncthreads();
    }
    if (tid < 4) {
        float invB = 1.f / (float)B;
        float mean = s_sum[0][tid] * invB;
        float var  = s_sq[0][tid] * invB - mean * mean;   // biased var
        float sc   = gamma[tid] * rsqrtf(var + 1e-5f);
        s_scale[tid] = sc;
        s_shift[tid] = beta[tid] - mean * sc;
    }
    __syncthreads();
    float sc[4], sh[4];
#pragma unroll
    for (int w = 0; w < 4; ++w) { sc[w] = s_scale[w]; sh[w] = s_shift[w]; }

    for (int r = tid; r < B; r += 256) {
        float4 v = reinterpret_cast<const float4*>(m4)[r];
        float mn[4] = { v.x * sc[0] + sh[0], v.y * sc[1] + sh[1],
                        v.z * sc[2] + sh[2], v.w * sc[3] + sh[3] };
        float logits[10], mx = -1e30f;
#pragma unroll
        for (int j = 0; j < 10; ++j) {
            float acc = bcls[j];
#pragma unroll
            for (int w = 0; w < 4; ++w) acc += mn[w] * wcls[j * 4 + w];
            logits[j] = acc;
            mx = fmaxf(mx, acc);
        }
        float se = 0.f;
#pragma unroll
        for (int j = 0; j < 10; ++j) se += __expf(logits[j] - mx);
        float lse = mx + __logf(se);
#pragma unroll
        for (int j = 0; j < 10; ++j) out[r * 10 + j] = logits[j] - lse;
    }
}

extern "C" void kernel_launch(void* const* d_in, const int* in_sizes, int n_in,
                              void* d_out, int out_size, void* d_ws, size_t ws_size,
                              hipStream_t stream) {
    const float* x       = (const float*)d_in[0];
    const float* uqf_re  = (const float*)d_in[1];
    const float* uqf_im  = (const float*)d_in[2];
    const float* uqc_re  = (const float*)d_in[3];
    const float* uqc_im  = (const float*)d_in[4];
    const float* gamma   = (const float*)d_in[5];
    const float* beta    = (const float*)d_in[6];
    const float* wcls    = (const float*)d_in[7];
    const float* bcls    = (const float*)d_in[8];
    float* out = (float*)d_out;

    int B = out_size / 10;
    float* m4 = (float*)d_ws;                    // B x 4 floats of scratch

    int tiles  = (B + 15) / 16;                  // 16 batch elements per wave
    int blocks = (tiles + 7) / 8;                // 8 waves (256 threads) per block
    quanv_qfc_kernel<<<blocks, 256, 0, stream>>>(x, uqf_re, uqf_im,
                                                 uqc_re, uqc_im, m4, B);
    bn_head_kernel<<<1, 256, 0, stream>>>(m4, gamma, beta, wcls, bcls, out, B);
}
